// Attention_44744969290295
// MI455X (gfx1250) — matrix operand order repaired and checked by reading, not compiled, not run on hardware
//
#include <hip/hip_runtime.h>
#include <hip/hip_bf16.h>
#include <math.h>

// ---------------------------------------------------------------------------
// Problem constants (fixed by the reference: B=2, S=2048, H=2048, 16 heads)
// ---------------------------------------------------------------------------
typedef __bf16 bf16_t;
typedef __attribute__((ext_vector_type(16))) __bf16 v16bf;
typedef __attribute__((ext_vector_type(8)))  __bf16 v8bf;
typedef __attribute__((ext_vector_type(8)))  float  v8f;
typedef __attribute__((ext_vector_type(4)))  int    v4i;

constexpr int BATCH = 2;
constexpr int SEQ   = 2048;
constexpr int HID   = 2048;
constexpr int NHEAD = 16;
constexpr int HD    = 128;                 // head dim
constexpr int MROWS = BATCH * SEQ;         // 4096
constexpr float SCALE = 0.022097086912079608f; // 1/(sqrt(128) * (LAYER_INDEX+1))

#define WMMA_BF16(a, b, c) \
  __builtin_amdgcn_wmma_f32_16x16x32_bf16(false, (a), false, (b), (short)0, (c), false, false)

// ---------------------------------------------------------------------------
// gfx1250 async global->LDS copy (ASYNCcnt-tracked), with sync fallback.
// ---------------------------------------------------------------------------
#ifndef __has_builtin
#define __has_builtin(x) 0
#endif

#if defined(__AMDGCN__) && \
    __has_builtin(__builtin_amdgcn_global_load_async_to_lds_b128) && \
    __has_builtin(__builtin_amdgcn_s_wait_asynccnt)
#define USE_ASYNC_LDS 1
#else
#define USE_ASYNC_LDS 0
#endif

#define AS_GLOBAL __attribute__((address_space(1)))
#define AS_LDS    __attribute__((address_space(3)))

__device__ __forceinline__ void cp_g2l_16B(const bf16_t* g, bf16_t* l) {
#if USE_ASYNC_LDS
  __builtin_amdgcn_global_load_async_to_lds_b128(
      (AS_GLOBAL v4i*)(g),
      (AS_LDS v4i*)(l),
      0, 0);
#else
  *(v8bf*)l = *(const v8bf*)g;
#endif
}

__device__ __forceinline__ void wait_async_copies() {
#if USE_ASYNC_LDS
  __builtin_amdgcn_s_wait_asynccnt(0);
#endif
}

// ---------------------------------------------------------------------------
// fp32 -> bf16 convert
// ---------------------------------------------------------------------------
__global__ void __launch_bounds__(256) cvt_f32_bf16(const float* __restrict__ in,
                                                    bf16_t* __restrict__ out, int n) {
  int i = blockIdx.x * blockDim.x + threadIdx.x;
  if (i < n) out[i] = (bf16_t)in[i];
}

// ---------------------------------------------------------------------------
// WMMA GEMM:  C[M,N] = A[M,K] * W[N,K]^T   (A,W bf16 row-major, K=HID fixed)
// MODE 0: store bf16 row-major (M,N)
// MODE 1: store fp32 row-major (M,N)
// MODE 2: store bf16 transposed per-head:  out[b,h,d,s]  (for V)
//
// Block = 128 threads (4 waves) computing a 16x256 strip.  K-chunks of 32 are
// double-buffered in LDS via async global->LDS copies: A tile 16x32 (1 KB) +
// B tile 256x32 (16 KB) per buffer, shared by all 4 waves.
// ---------------------------------------------------------------------------
template <int MODE>
__global__ void __launch_bounds__(128) gemm_bf16_wmma(const bf16_t* __restrict__ A,
                                                      const bf16_t* __restrict__ W,
                                                      void* __restrict__ Cout) {
  __shared__ alignas(64) bf16_t sA[2][16 * 32];
  __shared__ alignas(64) bf16_t sB[2][256 * 32];

  const int tid  = threadIdx.x;
  const int lane = tid & 31;
  const int wave = tid >> 5;
  const int hs   = lane >> 4;        // half-wave select
  const int col  = lane & 15;
  const int m0   = blockIdx.x * 16;
  const int n0b  = blockIdx.y * 256;
  const int n0   = n0b + wave * 64;

  // Stage one 32-wide K chunk into LDS buffer `buf` (async).
  auto stage = [&](int kk, int buf) {
    if (tid < 64) {                                   // A: 16 rows x 64 B
      const int row = tid >> 2, seg = tid & 3;
      cp_g2l_16B(A + (size_t)(m0 + row) * HID + kk + seg * 8,
                 &sA[buf][row * 32 + seg * 8]);
    }
#pragma unroll
    for (int i = 0; i < 8; ++i) {                     // B: 256 rows x 64 B
      const int sidx = tid * 8 + i;
      const int row = sidx >> 2, seg = sidx & 3;
      cp_g2l_16B(W + (size_t)(n0b + row) * HID + kk + seg * 8,
                 &sB[buf][row * 32 + seg * 8]);
    }
  };

  v8f c0 = {0.f,0.f,0.f,0.f,0.f,0.f,0.f,0.f};
  v8f c1 = c0, c2 = c0, c3 = c0;

  stage(0, 0);
  int buf = 0;
  for (int kk = 0; kk < HID; kk += 32) {
    wait_async_copies();        // this wave's staged chunk is in LDS
    __syncthreads();            // ... and every other wave's share too
    if (kk + 32 < HID) stage(kk + 32, buf ^ 1);   // overlap next chunk

    // A fragment (16x32 bf16): lane<16 -> K 0..15, lane>=16 -> K 16..31
    v8bf g0 = *(const v8bf*)&sA[buf][col * 32 + hs * 16];
    v8bf g1 = *(const v8bf*)&sA[buf][col * 32 + hs * 16 + 8];
    v16bf a = __builtin_shufflevector(g0, g1, 0,1,2,3,4,5,6,7,8,9,10,11,12,13,14,15);

    // B fragments: column n = row n of W; lane<16 K 0..15, lane>=16 K 16..31
    const int br = wave * 64 + col;
    v16bf b0 = *(const v16bf*)&sB[buf][(br +  0) * 32 + hs * 16];
    v16bf b1 = *(const v16bf*)&sB[buf][(br + 16) * 32 + hs * 16];
    v16bf b2 = *(const v16bf*)&sB[buf][(br + 32) * 32 + hs * 16];
    v16bf b3 = *(const v16bf*)&sB[buf][(br + 48) * 32 + hs * 16];
    c0 = WMMA_BF16(a, b0, c0);
    c1 = WMMA_BF16(a, b1, c1);
    c2 = WMMA_BF16(a, b2, c2);
    c3 = WMMA_BF16(a, b3, c3);

    buf ^= 1;
  }

  v8f cc[4] = {c0, c1, c2, c3};
#pragma unroll
  for (int t = 0; t < 4; ++t) {
#pragma unroll
    for (int r = 0; r < 8; ++r) {
      const int m = m0 + r + hs * 8;          // C layout: VGPR r -> row r / r+8
      const int n = n0 + t * 16 + col;
      const float v = cc[t][r];
      if (MODE == 0) {
        ((bf16_t*)Cout)[(size_t)m * HID + n] = (bf16_t)v;
      } else if (MODE == 1) {
        ((float*)Cout)[(size_t)m * HID + n] = v;
      } else {
        const int bb = m >> 11, s = m & (SEQ - 1);
        const int hh = n >> 7,  d = n & (HD - 1);
        ((bf16_t*)Cout)[(((size_t)bb * NHEAD + hh) * HD + d) * SEQ + s] = (bf16_t)v;
      }
    }
  }
}

// ---------------------------------------------------------------------------
// RoPE + reshape: (b,s,H) bf16 -> (b,h,s,d) bf16, rotation in fp32.
// One thread per (b,s,h,pair).
// ---------------------------------------------------------------------------
__global__ void __launch_bounds__(256) rope_kernel(const bf16_t* __restrict__ q0,
                                                   const bf16_t* __restrict__ k0,
                                                   const float* __restrict__ fcos,
                                                   const float* __restrict__ fsin,
                                                   bf16_t* __restrict__ qo,
                                                   bf16_t* __restrict__ ko) {
  const int idx = blockIdx.x * blockDim.x + threadIdx.x;  // 2*2048*16*64 = 2^22
  const int i = idx & 63;
  const int h = (idx >> 6) & (NHEAD - 1);
  const int s = (idx >> 10) & (SEQ - 1);
  const int b = idx >> 21;
  const size_t in  = ((size_t)b * SEQ + s) * HID + h * HD + 2 * i;
  const size_t out = (((size_t)b * NHEAD + h) * SEQ + s) * HD + 2 * i;
  const float c  = fcos[s * 64 + i];
  const float sn = fsin[s * 64 + i];
  const float q1 = (float)q0[in], q2 = (float)q0[in + 1];
  qo[out]     = (bf16_t)(q1 * c - q2 * sn);
  qo[out + 1] = (bf16_t)(q1 * sn + q2 * c);
  const float k1 = (float)k0[in], k2 = (float)k0[in + 1];
  ko[out]     = (bf16_t)(k1 * c - k2 * sn);
  ko[out + 1] = (bf16_t)(k1 * sn + k2 * c);
}

// ---------------------------------------------------------------------------
// Flash attention, causal, one wave per (b, h, 16-query tile).
//   Q,K: (b,h,s,128) bf16 (RoPE'd);  VT: (b,h,128,s) bf16;  mask: (b,1,S,S) f32
//   out: (b,s,H) bf16
// Key blocks of 32 so the P A-fragment has a full K=32.
// ---------------------------------------------------------------------------
__global__ void __launch_bounds__(32) flash_attn(const bf16_t* __restrict__ Q,
                                                 const bf16_t* __restrict__ K,
                                                 const bf16_t* __restrict__ VT,
                                                 const float* __restrict__ mask,
                                                 bf16_t* __restrict__ O) {
  __shared__ float sP[16][36];   // 16x32 P tile (+pad), used for layout transpose

  const int lane = threadIdx.x & 31;
  const int hs   = lane >> 4;
  const int col  = lane & 15;
  const int q0   = blockIdx.x * 16;
  const int h    = blockIdx.y;
  const int b    = blockIdx.z;

  const bf16_t* __restrict__ Qh = Q  + (((size_t)b * NHEAD + h) * SEQ) * HD;
  const bf16_t* __restrict__ Kh = K  + (((size_t)b * NHEAD + h) * SEQ) * HD;
  const bf16_t* __restrict__ Vh = VT + (((size_t)b * NHEAD + h) * HD) * SEQ;
  const float*  __restrict__ Mb = mask + (size_t)b * SEQ * SEQ;

  // Q A-fragments for the 4 K-chunks of head_dim=128
  v16bf qa[4];
  {
    const bf16_t* qr = Qh + (size_t)(q0 + col) * HD;
#pragma unroll
    for (int c4 = 0; c4 < 4; ++c4) {
      const bf16_t* p = qr + c4 * 32 + hs * 16;
      v8bf g0 = *(const v8bf*)p;
      v8bf g1 = *(const v8bf*)(p + 8);
      qa[c4] = __builtin_shufflevector(g0, g1, 0,1,2,3,4,5,6,7,8,9,10,11,12,13,14,15);
    }
  }

  v8f o[8];
#pragma unroll
  for (int t = 0; t < 8; ++t) o[t] = v8f{0.f,0.f,0.f,0.f,0.f,0.f,0.f,0.f};
  float mrow[8], lrow[8];
#pragma unroll
  for (int r = 0; r < 8; ++r) { mrow[r] = -INFINITY; lrow[r] = 0.f; }

  for (int jb = 0; jb < q0 + 16; jb += 32) {   // causal: only blocks touching keys <= q0+15
    // prefetch next key block's rows (streaming dimension)
    __builtin_prefetch(Kh + (size_t)(jb + 32 + col) * HD, 0, 3);

    // ---- S = Q K^T for 32 keys (two 16x16 tiles) ----
    v8f sl = {0.f,0.f,0.f,0.f,0.f,0.f,0.f,0.f};
    v8f sr = sl;
#pragma unroll
    for (int c4 = 0; c4 < 4; ++c4) {
      const size_t ko = (size_t)(c4 * 32 + hs * 16);
      v16bf bl = *(const v16bf*)(Kh + (size_t)(jb +      col) * HD + ko);
      v16bf br = *(const v16bf*)(Kh + (size_t)(jb + 16 + col) * HD + ko);
      sl = WMMA_BF16(qa[c4], bl, sl);
      sr = WMMA_BF16(qa[c4], br, sr);
    }

    // ---- scale + mask + causal + online softmax ----
#pragma unroll
    for (int r = 0; r < 8; ++r) {
      const int qrow = q0 + r + hs * 8;
      const int kl = jb + col, kr = kl + 16;
      float a = sl[r] * SCALE + Mb[(size_t)qrow * SEQ + kl] + (kl > qrow ? -1e30f : 0.f);
      float c = sr[r] * SCALE + Mb[(size_t)qrow * SEQ + kr] + (kr > qrow ? -1e30f : 0.f);

      float t = fmaxf(a, c);
      for (int off = 1; off < 16; off <<= 1) t = fmaxf(t, __shfl_xor(t, off, 32));
      const float mn = fmaxf(mrow[r], t);
      const float alpha = __expf(mrow[r] - mn);     // exp(-inf)=0 on first block
      mrow[r] = mn;

      const float pa = __expf(a - mn);
      const float pb = __expf(c - mn);
      float rs = pa + pb;
      for (int off = 1; off < 16; off <<= 1) rs += __shfl_xor(rs, off, 32);
      lrow[r] = lrow[r] * alpha + rs;

#pragma unroll
      for (int t8 = 0; t8 < 8; ++t8) o[t8][r] = o[t8][r] * alpha;

      const int row = r + hs * 8;                   // C layout -> LDS tile
      sP[row][col]      = pa;
      sP[row][col + 16] = pb;
    }

    // ---- repack P (16x32) into the 16-bit A-fragment layout via LDS ----
    v16bf pfrag;
#pragma unroll
    for (int e = 0; e < 8; ++e) {
      pfrag[e]     = (bf16_t)sP[col][hs * 16 + e];
      pfrag[e + 8] = (bf16_t)sP[col][hs * 16 + 8 + e];
    }

    // ---- O += P @ V  (8 n-tiles over head_dim) ----
#pragma unroll
    for (int t8 = 0; t8 < 8; ++t8) {
      const bf16_t* pv = Vh + (size_t)(t8 * 16 + col) * SEQ + jb + hs * 16;
      v16bf vb = *(const v16bf*)pv;
      o[t8] = WMMA_BF16(pfrag, vb, o[t8]);
    }
  }

  // ---- epilogue: divide by row sums, write (b,s,H) bf16 ----
#pragma unroll
  for (int r = 0; r < 8; ++r) {
    const int qrow = q0 + r + hs * 8;
    const float inv = 1.0f / lrow[r];
#pragma unroll
    for (int t8 = 0; t8 < 8; ++t8) {
      const int d = t8 * 16 + col;
      O[((size_t)b * SEQ + qrow) * HID + h * HD + d] = (bf16_t)(o[t8][r] * inv);
    }
  }
}

// ---------------------------------------------------------------------------
// Host-side orchestration
// ---------------------------------------------------------------------------
extern "C" void kernel_launch(void* const* d_in, const int* in_sizes, int n_in,
                              void* d_out, int out_size, void* d_ws, size_t ws_size,
                              hipStream_t stream) {
  const float* x    = (const float*)d_in[0];
  const float* fcos = (const float*)d_in[1];
  const float* fsin = (const float*)d_in[2];
  const float* mask = (const float*)d_in[3];
  const float* wq   = (const float*)d_in[4];
  const float* wk   = (const float*)d_in[5];
  const float* wv   = (const float*)d_in[6];
  const float* wo   = (const float*)d_in[7];
  float* out = (float*)d_out;

  // Workspace carve-out (~151 MB total)
  char* ws = (char*)d_ws;
  size_t off = 0;
  auto carve = [&](size_t bytes) -> void* {
    void* p = ws + off;
    off += (bytes + 255) & ~(size_t)255;
    return p;
  };
  const size_t XB_B = (size_t)MROWS * HID * sizeof(bf16_t);
  const size_t WB_B = (size_t)HID * HID * sizeof(bf16_t);
  bf16_t* xb  = (bf16_t*)carve(XB_B);
  bf16_t* wqb = (bf16_t*)carve(WB_B);
  bf16_t* wkb = (bf16_t*)carve(WB_B);
  bf16_t* wvb = (bf16_t*)carve(WB_B);
  bf16_t* wob = (bf16_t*)carve(WB_B);
  bf16_t* q0b = (bf16_t*)carve(XB_B);   // pre-RoPE q, (b,s,H)
  bf16_t* k0b = (bf16_t*)carve(XB_B);   // pre-RoPE k, (b,s,H)
  bf16_t* vt  = (bf16_t*)carve(XB_B);   // V transposed, (b,h,d,s)
  bf16_t* qb  = (bf16_t*)carve(XB_B);   // RoPE'd q, (b,h,s,d)
  bf16_t* kb  = (bf16_t*)carve(XB_B);   // RoPE'd k, (b,h,s,d)
  bf16_t* ob  = (bf16_t*)carve(XB_B);   // attention out, (b,s,H)

  // 1) fp32 -> bf16 converts
  const int nX = MROWS * HID;       // 8,388,608
  const int nW = HID * HID;         // 4,194,304
  cvt_f32_bf16<<<nX / 256, 256, 0, stream>>>(x,  xb,  nX);
  cvt_f32_bf16<<<nW / 256, 256, 0, stream>>>(wq, wqb, nW);
  cvt_f32_bf16<<<nW / 256, 256, 0, stream>>>(wk, wkb, nW);
  cvt_f32_bf16<<<nW / 256, 256, 0, stream>>>(wv, wvb, nW);
  cvt_f32_bf16<<<nW / 256, 256, 0, stream>>>(wo, wob, nW);

  // 2) QKV projections (WMMA bf16, async-LDS double-buffered)
  dim3 gg(MROWS / 16, HID / 256);
  gemm_bf16_wmma<0><<<gg, 128, 0, stream>>>(xb, wqb, (void*)q0b);
  gemm_bf16_wmma<0><<<gg, 128, 0, stream>>>(xb, wkb, (void*)k0b);
  gemm_bf16_wmma<2><<<gg, 128, 0, stream>>>(xb, wvb, (void*)vt);

  // 3) RoPE + reshape to (b,h,s,d)
  const int nR = BATCH * SEQ * NHEAD * 64;  // 4,194,304 pairs
  rope_kernel<<<nR / 256, 256, 0, stream>>>(q0b, k0b, fcos, fsin, qb, kb);

  // 4) causal flash attention (WMMA for QK^T and P@V)
  dim3 ga(SEQ / 16, NHEAD, BATCH);
  flash_attn<<<ga, 32, 0, stream>>>(qb, kb, vt, mask, ob);

  // 5) output projection, fp32 store to d_out
  gemm_bf16_wmma<1><<<gg, 128, 0, stream>>>(ob, wob, (void*)out);
}
